// VimBlock_12463995093386
// MI455X (gfx1250) — matrix operand order repaired and checked by reading, not compiled, not run on hardware
//
#include <hip/hip_runtime.h>
#include <hip/hip_bf16.h>

// ---------------------------------------------------------------------------
// Types for CDNA5 WMMA (wave32, 16x16x32 bf16 -> f32)
// ---------------------------------------------------------------------------
typedef float        v8f   __attribute__((ext_vector_type(8)));
typedef __bf16       v16bf __attribute__((ext_vector_type(16)));
typedef __bf16       v8bf  __attribute__((ext_vector_type(8)));
typedef unsigned int u32x4 __attribute__((ext_vector_type(4)));
typedef int          i32x4 __attribute__((ext_vector_type(4)));
typedef int          i32x8 __attribute__((ext_vector_type(8)));

union FragBF16 { v16bf f; v8bf h[2]; };

#if __has_builtin(__builtin_amdgcn_tensor_load_to_lds) && __has_builtin(__builtin_amdgcn_s_wait_tensorcnt)
#define USE_TDM 1
#else
#define USE_TDM 0
#endif

__device__ __forceinline__ __bf16 f2bf(float f) {
    unsigned u = __builtin_bit_cast(unsigned, f);
    unsigned r = (u + 0x7FFFu + ((u >> 16) & 1u)) >> 16;
    unsigned short s = (unsigned short)r;
    return __builtin_bit_cast(__bf16, s);
}
__device__ __forceinline__ float bf2f(__bf16 b) {
    unsigned short s = __builtin_bit_cast(unsigned short, b);
    unsigned u = ((unsigned)s) << 16;
    return __builtin_bit_cast(float, u);
}
__device__ __forceinline__ float siluf(float v) { return v / (1.f + expf(-v)); }

// ---------------------------------------------------------------------------
// TDM: issue a 2D bf16 tile load (tileRows x 32) global -> LDS with row
// padding matching LDS layout [rows][32+8]. D# per CDNA5 ISA ch.8.
// ---------------------------------------------------------------------------
__device__ __forceinline__ void tdm_load_tile(const __bf16* gsrc, unsigned lds_off,
                                              int rowLen, int nRows, int tileRows) {
#if USE_TDM
    unsigned long long ga = (unsigned long long)(size_t)gsrc;
    u32x4 g0;
    g0[0] = 1u;                                            // count=1, user descriptor
    g0[1] = lds_off;                                       // LDS byte address
    g0[2] = (unsigned)(ga & 0xFFFFFFFFu);                  // global_addr[31:0]
    g0[3] = (unsigned)((ga >> 32) & 0x01FFFFFFu) | (2u << 30);  // addr[56:32], type=2
    i32x8 g1;
    g1[0] = (1 << 16)      // data_size = 2 bytes (bf16)
          | (1 << 20)      // pad_enable
          | (3 << 22)      // pad_interval: 16 DWORDs (= 32 bf16 row)
          | (3 << 25);     // pad_amount:   4 DWORDs (= 8 bf16, matches +8 LDS pad)
    g1[1] = (rowLen & 0xFFFF) << 16;                            // tensor_dim0[15:0]
    g1[2] = ((rowLen >> 16) & 0xFFFF) | ((nRows & 0xFFFF) << 16);   // td0 hi | td1 lo
    g1[3] = ((nRows >> 16) & 0xFFFF) | (32 << 16);              // td1 hi | tile_dim0=32
    g1[4] = tileRows & 0xFFFF;                                  // tile_dim1
    g1[5] = rowLen;                                             // tensor_dim0_stride lo
    g1[6] = 0;                                                  // stride hi | td1_stride lo
    g1[7] = 0;
    i32x4 z4 = {0, 0, 0, 0};
#if defined(__clang_major__) && (__clang_major__ >= 23)
    i32x8 z8 = {0, 0, 0, 0, 0, 0, 0, 0};
    __builtin_amdgcn_tensor_load_to_lds(g0, g1, z4, z4, z8, 0);
#else
    __builtin_amdgcn_tensor_load_to_lds(g0, g1, z4, z4, 0);
#endif
#else
    (void)gsrc; (void)lds_off; (void)rowLen; (void)nRows; (void)tileRows;
#endif
}

// ---------------------------------------------------------------------------
// Weight convert + transpose: W[K][N] f32  ->  Wt[N][K] bf16
// ---------------------------------------------------------------------------
__global__ __launch_bounds__(256) void wt_bf16_k(const float* __restrict__ W,
                                                 __bf16* __restrict__ Wt,
                                                 int K, int N) {
    int idx = blockIdx.x * 256 + threadIdx.x;
    if (idx >= K * N) return;
    int n = idx % N;
    int k = idx / N;
    Wt[(size_t)n * K + k] = f2bf(W[idx]);
}

// ---------------------------------------------------------------------------
// LayerNorm over D=256, one row per block (256 threads), output bf16
// ---------------------------------------------------------------------------
__global__ __launch_bounds__(256) void layernorm_k(const float* __restrict__ x,
                                                   const float* __restrict__ g,
                                                   const float* __restrict__ b,
                                                   __bf16* __restrict__ xn) {
    const int Dd = 256;
    int row = blockIdx.x;
    int t = threadIdx.x;
    float v = x[(size_t)row * Dd + t];
    float a = v, q = v * v;
#pragma unroll
    for (int off = 16; off > 0; off >>= 1) {
        a += __shfl_down(a, off);
        q += __shfl_down(q, off);
    }
    __shared__ float s1[8], s2[8];
    int w = t >> 5, l = t & 31;
    if (l == 0) { s1[w] = a; s2[w] = q; }
    __syncthreads();
    if (w == 0) {
        float aa = (l < 8) ? s1[l] : 0.f;
        float qq = (l < 8) ? s2[l] : 0.f;
#pragma unroll
        for (int off = 4; off > 0; off >>= 1) {
            aa += __shfl_down(aa, off);
            qq += __shfl_down(qq, off);
        }
        if (l == 0) { s1[0] = aa; s2[0] = qq; }
    }
    __syncthreads();
    float mean = s1[0] * (1.f / Dd);
    float var  = s2[0] * (1.f / Dd) - mean * mean;
    float rinv = rsqrtf(var + 1e-5f);
    xn[(size_t)row * Dd + t] = f2bf((v - mean) * rinv * g[t] + b[t]);
}

// ---------------------------------------------------------------------------
// WMMA GEMM: out[M][N] = A[M][K] (bf16) x Bt[N][K]^T (bf16) + epilogue
// Block tile 128x128, K-step 32, 8 waves (2 row-slabs x 4 col-slabs),
// each wave computes a 64x32 tile (acc[4][2], 8 WMMA per K-step).
// TDM path: double-buffered LDS; wave 0 issues tensor_load_to_lds for
// tile s+1 while all waves run WMMAs on tile s (TENSORcnt<=2 keeps the
// current tile's pair resident, next pair in flight).
// MODE 0: +bias                  -> f32 out
// MODE 1: +bias                  -> bf16 out
// MODE 2: +bias +bias2, softplus -> f32 out   (Delta)
// MODE 3: +bias +residual        -> f32 out   (final projection)
// ---------------------------------------------------------------------------
#define GT_M 128
#define GT_N 128
#define GT_K 32

__device__ __forceinline__ void wmma_step(const __bf16 (*__restrict__ Asb)[GT_K + 8],
                                          const __bf16 (*__restrict__ Bsb)[GT_K + 8],
                                          int wm, int wn, int half, int lr,
                                          v8f acc[4][2]) {
    // A (16x32 bf16): lanes 0-15 row=lr, K=0..7|16..23 ; lanes 16-31 K=8..15|24..31
    FragBF16 fa[4], fb[2];
#pragma unroll
    for (int i = 0; i < 4; ++i) {
        int row = wm * 64 + i * 16 + lr;
        fa[i].h[0] = *(const v8bf*)&Asb[row][half * 8];
        fa[i].h[1] = *(const v8bf*)&Asb[row][half * 8 + 16];
    }
    // B (32x16 bf16): lanes 0-15 col=lr K=0..7|16..23 ; lanes 16-31 K=8..15|24..31
#pragma unroll
    for (int j = 0; j < 2; ++j) {
        int col = wn * 32 + j * 16 + lr;
        fb[j].h[0] = *(const v8bf*)&Bsb[col][half * 8];
        fb[j].h[1] = *(const v8bf*)&Bsb[col][half * 8 + 16];
    }
#pragma unroll
    for (int i = 0; i < 4; ++i)
#pragma unroll
        for (int j = 0; j < 2; ++j)
            acc[i][j] = __builtin_amdgcn_wmma_f32_16x16x32_bf16(
                false, fa[i].f, false, fb[j].f, (short)0, acc[i][j], false, false);
}

template <int MODE>
__global__ __launch_bounds__(256) void gemm_wmma(const __bf16* __restrict__ A,
                                                 const __bf16* __restrict__ Bt,
                                                 const float* __restrict__ bias,
                                                 const float* __restrict__ bias2,
                                                 const float* __restrict__ resid,
                                                 float* __restrict__ outF,
                                                 __bf16* __restrict__ outB,
                                                 int Mtot, int N, int K) {
#if USE_TDM
    __shared__ __bf16 As[2][GT_M][GT_K + 8];
    __shared__ __bf16 Bs[2][GT_N][GT_K + 8];
#else
    __shared__ __bf16 As[1][GT_M][GT_K + 8];
    __shared__ __bf16 Bs[1][GT_N][GT_K + 8];
#endif

    const int tid  = threadIdx.x;
    const int lane = tid & 31;
    const int wave = tid >> 5;   // 0..7
    const int wm   = wave & 1;   // 64-row slab
    const int wn   = wave >> 1;  // 32-col slab (0..3)
    const int bm   = blockIdx.x * GT_M;
    const int bn   = blockIdx.y * GT_N;
    const int half = lane >> 4;  // 0/1
    const int lr   = lane & 15;

    v8f acc[4][2] = {};

#if USE_TDM
    const unsigned as_off0 = (unsigned)(size_t)&As[0][0][0];
    const unsigned as_off1 = (unsigned)(size_t)&As[1][0][0];
    const unsigned bs_off0 = (unsigned)(size_t)&Bs[0][0][0];
    const unsigned bs_off1 = (unsigned)(size_t)&Bs[1][0][0];
    const int nSteps = K / GT_K;

    if (tid < 32) {  // prologue: tile 0 -> buffer 0
        tdm_load_tile(&A[(size_t)bm * K], as_off0, K, Mtot, GT_M);
        tdm_load_tile(&Bt[(size_t)bn * K], bs_off0, K, N, GT_N);
    }
    for (int s = 0; s < nSteps; ++s) {
        const int cur = s & 1;
        if (tid < 32) {
            if (s + 1 < nSteps) {  // issue next tile into the other buffer
                int k1 = (s + 1) * GT_K;
                tdm_load_tile(&A[(size_t)bm * K + k1], cur ? as_off0 : as_off1, K, Mtot, GT_M);
                tdm_load_tile(&Bt[(size_t)bn * K + k1], cur ? bs_off0 : bs_off1, K, N, GT_N);
                // 4 outstanding; <=2 left  ==> current tile's pair has landed
                __builtin_amdgcn_s_wait_tensorcnt(2);
            } else {
                __builtin_amdgcn_s_wait_tensorcnt(0);
            }
        }
        __syncthreads();
        wmma_step(As[cur], Bs[cur], wm, wn, half, lr, acc);
        __syncthreads();  // buffer safe to overwrite two steps later
    }
#else
    for (int k0 = 0; k0 < K; k0 += GT_K) {
        {
            int idx0 = tid * 8;
#pragma unroll
            for (int p = 0; p < 2; ++p) {
                int idx = idx0 + p * 2048;       // A tile: 128x32 = 4096 bf16
                int r = idx >> 5, c = idx & 31;
                *(v8bf*)&As[0][r][c] = *(const v8bf*)&A[(size_t)(bm + r) * K + k0 + c];
                *(v8bf*)&Bs[0][r][c] = *(const v8bf*)&Bt[(size_t)(bn + r) * K + k0 + c];
            }
            if (k0 + GT_K < K) {  // prefetch next K tile (global_prefetch_b8)
                int r = idx0 >> 5, c = idx0 & 31;
                __builtin_prefetch(&A[(size_t)(bm + r) * K + k0 + GT_K + c], 0, 0);
                __builtin_prefetch(&Bt[(size_t)(bn + r) * K + k0 + GT_K + c], 0, 0);
            }
        }
        __syncthreads();
        wmma_step(As[0], Bs[0], wm, wn, half, lr, acc);
        __syncthreads();
    }
#endif

    // --- epilogue: lane holds col n = lr, rows (half*8 + v) within each 16x16 ---
#pragma unroll
    for (int i = 0; i < 4; ++i) {
#pragma unroll
        for (int j = 0; j < 2; ++j) {
            int col = bn + wn * 32 + j * 16 + lr;
            float bv = bias ? bias[col] : 0.f;
            if (MODE == 2 && bias2) bv += bias2[col];
#pragma unroll
            for (int v = 0; v < 8; ++v) {
                int row = bm + wm * 64 + i * 16 + half * 8 + v;
                float val = acc[i][j][v] + bv;
                if (MODE == 2) val = (val > 20.f) ? val : log1pf(expf(val));  // softplus
                if (MODE == 3) val += resid[(size_t)row * N + col];
                if (MODE == 1) outB[(size_t)row * N + col] = f2bf(val);
                else           outF[(size_t)row * N + col] = val;
            }
        }
    }
}

// ---------------------------------------------------------------------------
// Depthwise conv (K=3, 'same') fwd + time-flipped bwd, + SiLU.
// Writes both bf16 (GEMM inputs) and f32 (SSM inputs).
// ---------------------------------------------------------------------------
__global__ __launch_bounds__(256) void conv_silu_k(const float* __restrict__ xp,
                                                   const float* __restrict__ wf,
                                                   const float* __restrict__ bf,
                                                   const float* __restrict__ wb,
                                                   const float* __restrict__ bb,
                                                   __bf16* __restrict__ xf_bf,
                                                   float* __restrict__ xf_f,
                                                   __bf16* __restrict__ xb_bf,
                                                   float* __restrict__ xb_f,
                                                   int Mlen, int Ee) {
    int idx = blockIdx.x * 256 + threadIdx.x;  // over R*E
    int e = idx % Ee;
    int row = idx / Ee;
    int m = row % Mlen;
    const float* base = xp + (size_t)(row - m) * Ee + e;
    float c = base[(size_t)m * Ee];
    float l = (m > 0)        ? base[(size_t)(m - 1) * Ee] : 0.f;
    float r = (m < Mlen - 1) ? base[(size_t)(m + 1) * Ee] : 0.f;
    float vf = wf[e * 3 + 0] * l + wf[e * 3 + 1] * c + wf[e * 3 + 2] * r + bf[e];
    float vb = wb[e * 3 + 0] * r + wb[e * 3 + 1] * c + wb[e * 3 + 2] * l + bb[e];
    vf = siluf(vf);
    vb = siluf(vb);
    xf_bf[idx] = f2bf(vf);  xf_f[idx] = vf;
    xb_bf[idx] = f2bf(vb);  xb_f[idx] = vb;
}

// ---------------------------------------------------------------------------
// SSM scan (log-space prefix product, faithful to reference formulation).
// One block = 16 e-channels x 16 n-states (256 thr), sequential over M.
// ---------------------------------------------------------------------------
__global__ __launch_bounds__(256) void ssm_k(const float* __restrict__ Delta,
                                             const float* __restrict__ xd,
                                             const __bf16* __restrict__ Bm,
                                             const __bf16* __restrict__ Cm,
                                             const float* __restrict__ Ap,
                                             float* __restrict__ y,
                                             int Mlen, int Ee, int Nn) {
    int n  = threadIdx.x & 15;
    int el = threadIdx.x >> 4;
    int nblk = Ee / 16;
    int eblk = blockIdx.x % nblk;
    int bb   = blockIdx.x / nblk;
    int e = eblk * 16 + el;
    float Aen = Ap[e * Nn + n];

    float la = 0.f, sgn = 1.f, S = 0.f;
    const float eps = 1e-6f;
    for (int m = 0; m < Mlen; ++m) {
        int row = bb * Mlen + m;
        float dlt = Delta[(size_t)row * Ee + e];
        float xv  = xd[(size_t)row * Ee + e];
        size_t bi = (size_t)row * Ee * Nn + (size_t)e * Nn + n;
        float Bv = bf2f(Bm[bi]);
        float Cv = bf2f(Cm[bi]);
        float Aa = dlt * Aen;
        float s  = (Aa > 0.f) ? 1.f : ((Aa < 0.f) ? -1.f : 0.f);
        sgn *= s;
        la  += logf(fmaxf(fabsf(Aa), eps));
        float P = sgn * expf(la);
        S += (1.f / (P + eps)) * (dlt * Bv) * xv;
        float h = P * S;
        float acc = h * Cv;
        acc += __shfl_xor(acc, 1);
        acc += __shfl_xor(acc, 2);
        acc += __shfl_xor(acc, 4);
        acc += __shfl_xor(acc, 8);
        if (n == 0) y[(size_t)row * Ee + e] = acc;
    }
}

// ---------------------------------------------------------------------------
// Gate: yg = (y_f + y_b) * silu(z)  -> bf16
// ---------------------------------------------------------------------------
__global__ __launch_bounds__(256) void gate_k(const float* __restrict__ yf,
                                              const float* __restrict__ yb,
                                              const float* __restrict__ z,
                                              __bf16* __restrict__ yg, int n) {
    int i = blockIdx.x * 256 + threadIdx.x;
    if (i >= n) return;
    float g = siluf(z[i]);
    yg[i] = f2bf((yf[i] + yb[i]) * g);
}

// ---------------------------------------------------------------------------
// Host-side orchestration
// ---------------------------------------------------------------------------
extern "C" void kernel_launch(void* const* d_in, const int* in_sizes, int n_in,
                              void* d_out, int out_size, void* d_ws, size_t ws_size,
                              hipStream_t stream) {
    (void)in_sizes; (void)n_in; (void)out_size; (void)ws_size;

    const int Bsz = 2, Mlen = 1024, Dd = 256, Ee = 512, Nn = 16;
    const int R  = Bsz * Mlen;   // 2048 rows
    const int EN = Ee * Nn;      // 8192

    const float* x    = (const float*)d_in[0];
    const float* ln_g = (const float*)d_in[1];
    const float* ln_b = (const float*)d_in[2];
    const float* Wx   = (const float*)d_in[3];
    const float* bx   = (const float*)d_in[4];
    const float* Wz   = (const float*)d_in[5];
    const float* bz   = (const float*)d_in[6];
    const float* cfw  = (const float*)d_in[7];
    const float* cfb  = (const float*)d_in[8];
    const float* cbw  = (const float*)d_in[9];
    const float* cbb  = (const float*)d_in[10];
    const float* WBf  = (const float*)d_in[11];
    const float* bBf  = (const float*)d_in[12];
    const float* WCf  = (const float*)d_in[13];
    const float* bCf  = (const float*)d_in[14];
    const float* WDf  = (const float*)d_in[15];
    const float* bDf  = (const float*)d_in[16];
    const float* Af   = (const float*)d_in[17];
    const float* dbf  = (const float*)d_in[18];
    const float* WBb  = (const float*)d_in[19];
    const float* bBb  = (const float*)d_in[20];
    const float* WCb  = (const float*)d_in[21];
    const float* bCb  = (const float*)d_in[22];
    const float* WDb  = (const float*)d_in[23];
    const float* bDb  = (const float*)d_in[24];
    const float* Ab   = (const float*)d_in[25];
    const float* dbb  = (const float*)d_in[26];
    const float* Wout = (const float*)d_in[27];
    const float* bout = (const float*)d_in[28];

    char* wsb = (char*)d_ws;
    size_t off = 0;
    auto take = [&](size_t bytes) -> void* {
        void* p = wsb + off;
        off += (bytes + 255) & ~(size_t)255;
        return p;
    };

    __bf16* xn    = (__bf16*)take((size_t)R * Dd * 2);
    __bf16* Wxt   = (__bf16*)take((size_t)Dd * Ee * 2);
    __bf16* Wzt   = (__bf16*)take((size_t)Dd * Ee * 2);
    __bf16* WBft  = (__bf16*)take((size_t)Ee * EN * 2);
    __bf16* WCft  = (__bf16*)take((size_t)Ee * EN * 2);
    __bf16* WBbt  = (__bf16*)take((size_t)Ee * EN * 2);
    __bf16* WCbt  = (__bf16*)take((size_t)Ee * EN * 2);
    __bf16* WDft  = (__bf16*)take((size_t)Ee * Ee * 2);
    __bf16* WDbt  = (__bf16*)take((size_t)Ee * Ee * 2);
    __bf16* Woutt = (__bf16*)take((size_t)Ee * Dd * 2);
    float*  xproj = (float*)take((size_t)R * Ee * 4);
    float*  zbuf  = (float*)take((size_t)R * Ee * 4);
    __bf16* xf_bf = (__bf16*)take((size_t)R * Ee * 2);
    float*  xf_f  = (float*)take((size_t)R * Ee * 4);
    __bf16* xb_bf = (__bf16*)take((size_t)R * Ee * 2);
    float*  xb_f  = (float*)take((size_t)R * Ee * 4);
    float*  DltF  = (float*)take((size_t)R * Ee * 4);
    float*  DltB  = (float*)take((size_t)R * Ee * 4);
    __bf16* BmB   = (__bf16*)take((size_t)R * EN * 2);   // reused fwd then bwd
    __bf16* CmB   = (__bf16*)take((size_t)R * EN * 2);
    float*  y_f   = (float*)take((size_t)R * Ee * 4);
    float*  y_b   = (float*)take((size_t)R * Ee * 4);
    __bf16* yg    = (__bf16*)take((size_t)R * Ee * 2);

    auto wt = [&](const float* W, __bf16* Wt, int K, int N) {
        int total = K * N;
        wt_bf16_k<<<(total + 255) / 256, 256, 0, stream>>>(W, Wt, K, N);
    };
    wt(Wx, Wxt, Dd, Ee);
    wt(Wz, Wzt, Dd, Ee);
    wt(WBf, WBft, Ee, EN);
    wt(WCf, WCft, Ee, EN);
    wt(WBb, WBbt, Ee, EN);
    wt(WCb, WCbt, Ee, EN);
    wt(WDf, WDft, Ee, Ee);
    wt(WDb, WDbt, Ee, Ee);
    wt(Wout, Woutt, Ee, Dd);

    layernorm_k<<<R, 256, 0, stream>>>(x, ln_g, ln_b, xn);

    gemm_wmma<0><<<dim3(R / GT_M, Ee / GT_N), 256, 0, stream>>>(
        xn, Wxt, bx, nullptr, nullptr, xproj, nullptr, R, Ee, Dd);
    gemm_wmma<0><<<dim3(R / GT_M, Ee / GT_N), 256, 0, stream>>>(
        xn, Wzt, bz, nullptr, nullptr, zbuf, nullptr, R, Ee, Dd);

    conv_silu_k<<<(R * Ee) / 256, 256, 0, stream>>>(
        xproj, cfw, cfb, cbw, cbb, xf_bf, xf_f, xb_bf, xb_f, Mlen, Ee);

    // forward branch
    gemm_wmma<1><<<dim3(R / GT_M, EN / GT_N), 256, 0, stream>>>(
        xf_bf, WBft, bBf, nullptr, nullptr, nullptr, BmB, R, EN, Ee);
    gemm_wmma<1><<<dim3(R / GT_M, EN / GT_N), 256, 0, stream>>>(
        xf_bf, WCft, bCf, nullptr, nullptr, nullptr, CmB, R, EN, Ee);
    gemm_wmma<2><<<dim3(R / GT_M, Ee / GT_N), 256, 0, stream>>>(
        xf_bf, WDft, bDf, dbf, nullptr, DltF, nullptr, R, Ee, Ee);
    ssm_k<<<Bsz * (Ee / 16), 256, 0, stream>>>(DltF, xf_f, BmB, CmB, Af, y_f, Mlen, Ee, Nn);

    // backward branch (reuses Bm/Cm buffers)
    gemm_wmma<1><<<dim3(R / GT_M, EN / GT_N), 256, 0, stream>>>(
        xb_bf, WBbt, bBb, nullptr, nullptr, nullptr, BmB, R, EN, Ee);
    gemm_wmma<1><<<dim3(R / GT_M, EN / GT_N), 256, 0, stream>>>(
        xb_bf, WCbt, bCb, nullptr, nullptr, nullptr, CmB, R, EN, Ee);
    gemm_wmma<2><<<dim3(R / GT_M, Ee / GT_N), 256, 0, stream>>>(
        xb_bf, WDbt, bDb, dbb, nullptr, DltB, nullptr, R, Ee, Ee);
    ssm_k<<<Bsz * (Ee / 16), 256, 0, stream>>>(DltB, xb_f, BmB, CmB, Ab, y_b, Mlen, Ee, Nn);

    // gate + output projection with residual
    gate_k<<<(R * Ee) / 256, 256, 0, stream>>>(y_f, y_b, zbuf, yg, R * Ee);
    gemm_wmma<3><<<dim3(R / GT_M, Dd / GT_N), 256, 0, stream>>>(
        yg, Woutt, bout, nullptr, x, (float*)d_out, nullptr, R, Dd, Ee);
}
